// TrajectoryAttentionWM_21380347199595
// MI455X (gfx1250) — matrix operand order repaired and checked by reading, not compiled, not run on hardware
//
#include <hip/hip_runtime.h>
#include <hip/hip_bf16.h>
#include <stdint.h>

typedef __attribute__((ext_vector_type(16))) _Float16 v16h;
typedef __attribute__((ext_vector_type(8)))  _Float16 v8h;
typedef __attribute__((ext_vector_type(8)))  float    v8f;

#define BATCH 32
#define HH 256
#define WW 256
#define HW (HH*WW)
#define NPIX (BATCH*HW)      // 2097152
#define STEPS 15
#define R_    3.9f
#define EPS_  0.3f
#define BETA_ 0.15f
#define CLO   1.0e-4f
#define CHI   (1.0f - 1.0e-4f)

// stencil tiling
#define TIN  32
#define HALO 15
#define TOUT (TIN + 2*HALO)   // 62
#define TSTR 64               // padded LDS row stride

// ---------------------------------------------------------------------------
// Kernel 0: fold weights.
//   nca_input channels are affine in planes (x, last, S):
//     ch0=x, ch1=last, ch2=last-x, ch3..5 = Wv[v]*S + bv[v]
//   WB1[n=32][k=64] f16: k 0..26 = dil-1 taps (3 planes x 9), 27..53 = dil-2
//   taps (alpha folded), 54..63 = 0.  biasEff folds bp1, alpha*bp2, bv terms.
//   WB2[n=32][k=32] f16 = Wu1.
// ---------------------------------------------------------------------------
__global__ void prep_kernel(
    const float* __restrict__ Wv,  const float* __restrict__ bv,
    const float* __restrict__ Wp1, const float* __restrict__ bp1,
    const float* __restrict__ Wp2, const float* __restrict__ bp2,
    const float* __restrict__ alpha, const float* __restrict__ Wu1,
    const float* __restrict__ bu1, const float* __restrict__ Wu2,
    const float* __restrict__ bu2,
    _Float16* __restrict__ WB1, _Float16* __restrict__ WB2,
    float* __restrict__ biasEff, float* __restrict__ bu1o,
    float* __restrict__ wu2o, float* __restrict__ bu2o)
{
    int o = threadIdx.x;
    if (o >= 32) return;
    const float al = alpha[o];
    const float* w1 = Wp1 + (size_t)o * 54;   // [6][3][3]
    const float* w2 = Wp2 + (size_t)o * 54;
    float bacc = bp1[o] + al * bp2[o];
    for (int tap = 0; tap < 9; ++tap) {
        float p1x = w1[0*9+tap] - w1[2*9+tap];
        float p1l = w1[1*9+tap] + w1[2*9+tap];
        float p1s = Wv[0]*w1[3*9+tap] + Wv[1]*w1[4*9+tap] + Wv[2]*w1[5*9+tap];
        float p2x = (w2[0*9+tap] - w2[2*9+tap]) * al;
        float p2l = (w2[1*9+tap] + w2[2*9+tap]) * al;
        float p2s = (Wv[0]*w2[3*9+tap] + Wv[1]*w2[4*9+tap] + Wv[2]*w2[5*9+tap]) * al;
        WB1[o*64 + 0*9 + tap]      = (_Float16)p1x;
        WB1[o*64 + 1*9 + tap]      = (_Float16)p1l;
        WB1[o*64 + 2*9 + tap]      = (_Float16)p1s;
        WB1[o*64 + 27 + 0*9 + tap] = (_Float16)p2x;
        WB1[o*64 + 27 + 1*9 + tap] = (_Float16)p2l;
        WB1[o*64 + 27 + 2*9 + tap] = (_Float16)p2s;
        bacc += bv[0]*(w1[3*9+tap] + al*w2[3*9+tap])
              + bv[1]*(w1[4*9+tap] + al*w2[4*9+tap])
              + bv[2]*(w1[5*9+tap] + al*w2[5*9+tap]);
    }
    for (int k = 54; k < 64; ++k) WB1[o*64 + k] = (_Float16)0.0f;
    biasEff[o] = bacc;
    for (int k = 0; k < 32; ++k) WB2[o*32 + k] = (_Float16)Wu1[o*32 + k];
    bu1o[o] = bu1[o];
    wu2o[o] = Wu2[o];
    if (o == 0) bu2o[0] = bu2[0];
}

// ---------------------------------------------------------------------------
// Kernel 1: 15-step stencil + collapsed attention (online softmax).
// One block = 32x32 interior tile with halo 15 (62x62 in LDS, stride 64).
// Writes plane last (g_15) and plane S = sum_m softmax(a*g_m)*g_m.
// ---------------------------------------------------------------------------
__global__ __launch_bounds__(256) void traj_kernel(
    const float* __restrict__ x, const float* __restrict__ Klocal,
    const float* __restrict__ Wq, const float* __restrict__ bq,
    const float* __restrict__ Wk,
    float* __restrict__ lastP, float* __restrict__ SP)
{
    __shared__ float gbuf[TOUT*TSTR];
    __shared__ float mbuf[TOUT*TSTR];
    __shared__ float dbuf[TOUT*TSTR];

    const int tid = threadIdx.x;
    const int bx = blockIdx.x, by = blockIdx.y, b = blockIdx.z;
    const int gx0 = bx*TIN - HALO, gy0 = by*TIN - HALO;
    const float* xb = x + (size_t)b * HW;

    float Kw[9];
    #pragma unroll
    for (int j = 0; j < 9; ++j) Kw[j] = Klocal[j];
    const float qa = Wq[0]*Wk[0] + Wq[1]*Wk[1] + Wq[2]*Wk[2];
    const float qb = bq[0]*Wk[0] + bq[1]*Wk[1] + bq[2]*Wk[2];
    const float rsqrt3 = 0.5773502691896258f;

    for (int idx = tid; idx < TOUT*TOUT; idx += 256) {
        int cy = idx / TOUT, cx = idx % TOUT;
        int gy = gy0 + cy, gx = gx0 + cx;
        float v = 0.0f;
        if (gy >= 0 && gy < HH && gx >= 0 && gx < WW) v = xb[gy*WW + gx];
        gbuf[cy*TSTR + cx] = v;
        dbuf[cy*TSTR + cx] = v;
    }
    __syncthreads();

    // 4 interior pixels per thread: cid = tid + j*256 over 32x32
    float accM[4], accZ[4], accS[4], aC[4];
    #pragma unroll
    for (int j = 0; j < 4; ++j) {
        int cid = tid + j*256;
        int iy = cid >> 5, ix = cid & 31;
        float d = dbuf[(iy+HALO)*TSTR + (ix+HALO)];
        aC[j]  = (qa*d + qb) * rsqrt3;
        accM[j] = -3.0e38f; accZ[j] = 0.0f; accS[j] = 0.0f;
    }

    for (int t = 0; t < STEPS; ++t) {
        // pass 1: mapped = R*g*(1-g) (zero outside the image = conv zero-pad)
        for (int idx = tid; idx < TOUT*TOUT; idx += 256) {
            int cy = idx / TOUT, cx = idx % TOUT;
            int gy = gy0 + cy, gx = gx0 + cx;
            float g = gbuf[cy*TSTR + cx];
            float m = (gy >= 0 && gy < HH && gx >= 0 && gx < WW)
                      ? (R_ * g * (1.0f - g)) : 0.0f;
            mbuf[cy*TSTR + cx] = m;
        }
        __syncthreads();
        // pass 2: conv + blend + clamp (valid region shrinks by 1/step; halo 15 covers it)
        for (int idx = tid; idx < (TOUT-2)*(TOUT-2); idx += 256) {
            int cy = 1 + idx/(TOUT-2), cx = 1 + idx%(TOUT-2);
            const float* mc = &mbuf[cy*TSTR + cx];
            float local = Kw[0]*mc[-TSTR-1] + Kw[1]*mc[-TSTR] + Kw[2]*mc[-TSTR+1]
                        + Kw[3]*mc[-1]      + Kw[4]*mc[0]     + Kw[5]*mc[1]
                        + Kw[6]*mc[TSTR-1]  + Kw[7]*mc[TSTR]  + Kw[8]*mc[TSTR+1];
            float phys = (1.0f-EPS_)*mc[0] + EPS_*local;
            float g = (1.0f-BETA_)*phys + BETA_*dbuf[cy*TSTR + cx];
            g = fminf(fmaxf(g, CLO), CHI);
            gbuf[cy*TSTR + cx] = g;
        }
        __syncthreads();
        // online softmax accumulate over the interior
        #pragma unroll
        for (int j = 0; j < 4; ++j) {
            int cid = tid + j*256;
            int iy = cid >> 5, ix = cid & 31;
            float g = gbuf[(iy+HALO)*TSTR + (ix+HALO)];
            float s = aC[j] * g;
            float nM = fmaxf(accM[j], s);
            float e0 = __expf(accM[j] - nM);
            float e1 = __expf(s - nM);
            accZ[j] = accZ[j]*e0 + e1;
            accS[j] = accS[j]*e0 + e1*g;
            accM[j] = nM;
        }
    }

    #pragma unroll
    for (int j = 0; j < 4; ++j) {
        int cid = tid + j*256;
        int iy = cid >> 5, ix = cid & 31;
        int gy = by*TIN + iy, gx = bx*TIN + ix;
        size_t o = (size_t)b*HW + (size_t)gy*WW + gx;
        lastP[o] = gbuf[(iy+HALO)*TSTR + (ix+HALO)];
        SP[o]    = accS[j] / accZ[j];
    }
}

// ---------------------------------------------------------------------------
// Kernel 2: WMMA head. 128 pixels/block (8 waves x 16 pixels).
// GEMM1: im2col(3 planes x 18 taps, K=64 padded) x WB1 -> feat[16x32]
// GEMM2: relu(feat) x Wu1 -> h2[16x32]; scalar Wu2 dot; out = clip(last+corr).
// A-frag order used here (16-bit, 16x32): lane<16 holds K 0..15,
// lane>=16 holds K 16..31; B-frag: half-wave = 16 contiguous K.
// ---------------------------------------------------------------------------
__device__ __forceinline__ v16h load_a_frag(const _Float16* row, int hi) {
    v8h lo = *(const v8h*)(row + hi*16);
    v8h up = *(const v8h*)(row + hi*16 + 8);
    v16h a;
    #pragma unroll
    for (int i = 0; i < 8; ++i) { a[i] = lo[i]; a[8+i] = up[i]; }
    return a;
}

__global__ __launch_bounds__(256) void head_kernel(
    const float* __restrict__ xP, const float* __restrict__ lastP,
    const float* __restrict__ SP,
    const _Float16* __restrict__ WB1g, const _Float16* __restrict__ WB2g,
    const float* __restrict__ biasEffg, const float* __restrict__ bu1g,
    const float* __restrict__ wu2g, const float* __restrict__ bu2g,
    float* __restrict__ out)
{
    __shared__ __align__(32) _Float16 featLDS[128][64];
    __shared__ __align__(32) _Float16 hLDS[128][32];
    __shared__ float h2LDS[128][32];
    __shared__ __align__(32) _Float16 wb1[32][64];
    __shared__ __align__(32) _Float16 wb2[32][32];
    __shared__ float biasEff[32], bu1s[32], wu2s[32], bu2s;

    const int tid = threadIdx.x;
    {   // cooperative weight staging
        const uint32_t* s1 = (const uint32_t*)WB1g;
        uint32_t* d1 = (uint32_t*)&wb1[0][0];
        for (int i = tid; i < 1024; i += 256) d1[i] = s1[i];
        const uint32_t* s2 = (const uint32_t*)WB2g;
        uint32_t* d2 = (uint32_t*)&wb2[0][0];
        for (int i = tid; i < 512; i += 256) d2[i] = s2[i];
        if (tid < 32) { biasEff[tid] = biasEffg[tid]; bu1s[tid] = bu1g[tid]; wu2s[tid] = wu2g[tid]; }
        if (tid == 0) bu2s = bu2g[0];
    }

    // im2col gather: thread (p = tid>>1, half = tid&1) fills 32 K entries
    const int p    = tid >> 1;
    const int half = tid & 1;
    const int gp   = blockIdx.x * 128 + p;
    const int bImg = gp / HW;
    const int rem  = gp % HW;
    const int py   = rem / WW;
    const int px   = rem % WW;
    const size_t bOff = (size_t)bImg * HW;
    #pragma unroll
    for (int kk = 0; kk < 32; ++kk) {
        int k = half*32 + kk;
        float v = 0.0f;
        if (k < 54) {
            int kd  = (k < 27) ? k : (k - 27);
            int dil = (k < 27) ? 1 : 2;
            int c = kd / 9, tap = kd % 9;
            int ny = py + (tap/3 - 1)*dil;
            int nx = px + (tap%3 - 1)*dil;
            if (ny >= 0 && ny < HH && nx >= 0 && nx < WW) {
                const float* pl = (c == 0) ? xP : ((c == 1) ? lastP : SP);
                v = pl[bOff + (size_t)ny*WW + nx];
            }
        }
        featLDS[p][k] = (_Float16)v;
    }
    __syncthreads();

    const int lane = tid & 31;
    const int wv   = tid >> 5;
    const int pixBase = wv * 16;
    const int rowA = pixBase + (lane & 15);
    const int hi   = lane >> 4;
    const int n0   = lane & 15;

    // GEMM1: feat = im2col x WB1  (K=64 as 2x32, N=32 as 2x16)
    v8f c0 = {}; v8f c1 = {};
    #pragma unroll
    for (int kc = 0; kc < 2; ++kc) {
        v16h a  = load_a_frag(&featLDS[rowA][kc*32], hi);
        v16h b0 = *(const v16h*)&wb1[n0][kc*32 + hi*16];
        v16h b1 = *(const v16h*)&wb1[16 + n0][kc*32 + hi*16];
        c0 = __builtin_amdgcn_wmma_f32_16x16x32_f16(false, a, false, b0, (short)0, c0, false, false);
        c1 = __builtin_amdgcn_wmma_f32_16x16x32_f16(false, a, false, b1, (short)0, c1, false, false);
    }
    {   // bias + relu -> hLDS (D layout: M = r + 8*hi, N = lane&15)
        float bb0 = biasEff[n0], bb1 = biasEff[16 + n0];
        #pragma unroll
        for (int r = 0; r < 8; ++r) {
            int pix = pixBase + r + hi*8;
            hLDS[pix][n0]      = (_Float16)fmaxf(c0[r] + bb0, 0.0f);
            hLDS[pix][16 + n0] = (_Float16)fmaxf(c1[r] + bb1, 0.0f);
        }
    }
    __syncthreads();

    // GEMM2: h2 = relu(feat) x Wu1^T (K=32, N=32)
    v8f d0 = {}; v8f d1 = {};
    {
        v16h a  = load_a_frag(&hLDS[rowA][0], hi);
        v16h b0 = *(const v16h*)&wb2[n0][hi*16];
        v16h b1 = *(const v16h*)&wb2[16 + n0][hi*16];
        d0 = __builtin_amdgcn_wmma_f32_16x16x32_f16(false, a, false, b0, (short)0, d0, false, false);
        d1 = __builtin_amdgcn_wmma_f32_16x16x32_f16(false, a, false, b1, (short)0, d1, false, false);
    }
    {
        float bb0 = bu1s[n0], bb1 = bu1s[16 + n0];
        #pragma unroll
        for (int r = 0; r < 8; ++r) {
            int pix = pixBase + r + hi*8;
            h2LDS[pix][n0]      = fmaxf(d0[r] + bb0, 0.0f);
            h2LDS[pix][16 + n0] = fmaxf(d1[r] + bb1, 0.0f);
        }
    }
    __syncthreads();

    // Wu2 dot + residual + clip
    if (tid < 128) {
        float acc = bu2s;
        #pragma unroll
        for (int o = 0; o < 32; ++o) acc += wu2s[o] * h2LDS[tid][o];
        int g2 = blockIdx.x * 128 + tid;
        float res = lastP[g2] + acc;
        out[g2] = fminf(fmaxf(res, 0.0f), 1.0f);
    }
}

// ---------------------------------------------------------------------------
// Workspace layout: [lastP: NPIX f32][SP: NPIX f32][WB1 2048 f16][WB2 1024 f16]
//                   [biasEff 32 f32][bu1 32 f32][wu2 32 f32][bu2 1 f32]
// needs ~16 MB + 7 KB of ws.
// ---------------------------------------------------------------------------
extern "C" void kernel_launch(void* const* d_in, const int* in_sizes, int n_in,
                              void* d_out, int out_size, void* d_ws, size_t ws_size,
                              hipStream_t stream)
{
    const float* x     = (const float*)d_in[0];
    const float* Kl    = (const float*)d_in[1];
    const float* Wq    = (const float*)d_in[2];
    const float* bq    = (const float*)d_in[3];
    const float* Wk    = (const float*)d_in[4];
    // d_in[5] = bk: cancels inside softmax, unused
    const float* Wv    = (const float*)d_in[6];
    const float* bv    = (const float*)d_in[7];
    const float* Wp1   = (const float*)d_in[8];
    const float* bp1   = (const float*)d_in[9];
    const float* Wp2   = (const float*)d_in[10];
    const float* bp2   = (const float*)d_in[11];
    const float* alpha = (const float*)d_in[12];
    const float* Wu1   = (const float*)d_in[13];
    const float* bu1   = (const float*)d_in[14];
    const float* Wu2   = (const float*)d_in[15];
    const float* bu2   = (const float*)d_in[16];

    float* lastP = (float*)d_ws;
    float* SP    = lastP + NPIX;
    _Float16* WB1 = (_Float16*)(SP + NPIX);
    _Float16* WB2 = WB1 + 32*64;
    float* biasEff = (float*)(WB2 + 32*32);
    float* bu1o = biasEff + 32;
    float* wu2o = bu1o + 32;
    float* bu2o = wu2o + 32;

    prep_kernel<<<1, 32, 0, stream>>>(Wv, bv, Wp1, bp1, Wp2, bp2, alpha,
                                      Wu1, bu1, Wu2, bu2,
                                      WB1, WB2, biasEff, bu1o, wu2o, bu2o);

    dim3 gA(WW/TIN, HH/TIN, BATCH);   // (8,8,32)
    traj_kernel<<<gA, 256, 0, stream>>>(x, Kl, Wq, bq, Wk, lastP, SP);

    head_kernel<<<NPIX/128, 256, 0, stream>>>(x, lastP, SP, WB1, WB2,
                                              biasEff, bu1o, wu2o, bu2o,
                                              (float*)d_out);
}
